// RingDilatedAttentionV2_72438918414792
// MI455X (gfx1250) — matrix operand order repaired and checked by reading, not compiled, not run on hardware
//
#include <hip/hip_runtime.h>

// MI455X / gfx1250 flash-style dilated attention, S^T formulation + f16 prepack.
// Groups: g=0 heads0-3 sl=1024 rate=1; g=1 heads4-7 sl=2048 rate=2 off=1;
//         g=2 heads8-11 sl=4096 rate=4 off=2.
// Gather (all groups): gidx(j) = ((j & 1023) << g) + g  -- only 1024 distinct
// gathered keys per (head, segment), so K/V are staged once into d_ws as f16:
//   KG : [g][seg][head_in_g][j<1024][dim<64]   (A-layout-friendly, row-major)
//   VTG: [g][seg][head_in_g][dim<64][j<1024]   (transposed for the PV GEMM)
// Inner loop uses only global_load_b128 of packed halves; causal masking is
// confined to a single diagonal-block epilogue (one masked block per Q tile).

typedef __attribute__((ext_vector_type(16))) _Float16 v16h;
typedef __attribute__((ext_vector_type(8)))  _Float16 v8h;
typedef __attribute__((ext_vector_type(8)))  float    v8f;

namespace {
constexpr int kH   = 12;
constexpr int kD   = 64;
constexpr int kRow = kH * kD;              // 768 floats per sequence position

// per-group tile = 1024*64 halves = 65536; tiles: g0 4seg*4h=16, g1 8, g2 4
constexpr unsigned kTile   = 65536;        // halves per (seg,head) tile
constexpr unsigned kOffG1  = 16u * kTile;  // 1048576
constexpr unsigned kOffG2  = kOffG1 + 8u * kTile;  // 1572864
constexpr unsigned kKGHalves = kOffG2 + 4u * kTile; // 1835008 halves = 3.5MB
constexpr unsigned kWsHalves = 2u * kKGHalves;      // KG + VTG = 7MB
constexpr unsigned kPairsPerSide = kKGHalves / 2;   // 917504 element-pairs
}

union U16h { v16h v; v8h h[2]; };

// ---------------------------------------------------------------------------
// Prepack: gather + f16-convert K, and gather + transpose + f16-convert V.
// One thread handles one pair of halves. Grid covers exactly 2*917504 ids.
// ---------------------------------------------------------------------------
__global__ __launch_bounds__(256, 1) void prepack_kv(
    const float* __restrict__ K,
    const float* __restrict__ V,
    _Float16* __restrict__ ws)
{
  const unsigned id  = blockIdx.x * 256u + threadIdx.x;
  const bool isV = id >= kPairsPerSide;
  unsigned local = isV ? id - kPairsPerSide : id;

  int g; unsigned baseHalves;
  if (local < 524288u)      { g = 0; baseHalves = 0; }
  else if (local < 786432u) { g = 1; baseHalves = kOffG1; local -= 524288u; }
  else                      { g = 2; baseHalves = kOffG2; local -= 786432u; }

  const int sh   = (int)(local >> 15);     // seg*4 + head_in_group
  const int head = (g << 2) + (sh & 3);
  const int seg  = sh >> 2;
  const int sl   = 1024 << g;
  const int segb = seg * sl;
  const int hoff = head * kD;
  const unsigned tile = baseHalves + (unsigned)sh * kTile;

  if (!isV) {
    const int dim = (int)(local & 31) * 2;          // dim pair (contiguous)
    const int j   = (int)(local >> 5) & 1023;
    const int row = segb + (j << g) + g;            // dilated gather
    const float2 f = *(const float2*)(K + row * kRow + hoff + dim);
    _Float16* dst = ws + tile + (unsigned)j * 64u + (unsigned)dim;
    dst[0] = (_Float16)f.x;
    dst[1] = (_Float16)f.y;
  } else {
    const int j0  = (int)(local & 511) * 2;         // j pair (contiguous)
    const int dim = (int)(local >> 9) & 63;
    const int r0  = segb + (j0 << g) + g;
    const int r1  = segb + ((j0 + 1) << g) + g;
    _Float16* dst = ws + kKGHalves + tile + (unsigned)dim * 1024u + (unsigned)j0;
    dst[0] = (_Float16)V[r0 * kRow + hoff + dim];
    dst[1] = (_Float16)V[r1 * kRow + hoff + dim];
  }
}

// ---------------------------------------------------------------------------
// One flash-attention step over a 32-key block (packed f16 path).
// MASKED=true only for the single diagonal block of a causal Q tile.
// ---------------------------------------------------------------------------
template <bool MASKED>
__device__ __forceinline__ void flash_step_packed(
    const _Float16* __restrict__ kg, const _Float16* __restrict__ vt,
    int kb, int nl, int hi, int qrow,
    const v16h (&bq)[2], v8f (&oacc)[4], float& m_run, float& l_run)
{
  // ---- S^T = K x Q^T, 32 gathered keys (two 16x16 tiles) ----
  const int jbase = (kb * 32) & 1023;
  const _Float16* kp = kg + (unsigned)(jbase + nl) * 64u + 8u * (unsigned)hi;
  float pv[2][8];
#pragma unroll
  for (int t = 0; t < 2; ++t) {
    v8f s;
#pragma unroll
    for (int r = 0; r < 8; ++r) s[r] = 0.0f;
#pragma unroll
    for (int c = 0; c < 2; ++c) {
      U16h ak;
      ak.h[0] = *(const v8h*)(kp + t * 1024 + c * 32);
      ak.h[1] = *(const v8h*)(kp + t * 1024 + c * 32 + 16);
      s = __builtin_amdgcn_wmma_f32_16x16x32_f16(false, ak.v, false, bq[c],
                                                 (short)0, s, false, false);
    }
#pragma unroll
    for (int r = 0; r < 8; ++r) {
      float val = s[r] * 0.125f;                       // 1/sqrt(64)
      if (MASKED) {
        const int key = kb * 32 + t * 16 + r + 8 * hi; // gathered coordinate
        if (key > qrow) val = -1.0e30f;                // finite mask sentinel
      }
      pv[t][r] = val;
    }
  }

  // ---- online softmax: per-lane row + one partner shuffle ----
  float mx = pv[0][0];
#pragma unroll
  for (int r = 1; r < 8; ++r) mx = fmaxf(mx, pv[0][r]);
#pragma unroll
  for (int r = 0; r < 8; ++r) mx = fmaxf(mx, pv[1][r]);
  mx = fmaxf(mx, __shfl_xor(mx, 16, 32));
  const float mnew  = fmaxf(m_run, mx);
  const float alpha = __expf(m_run - mnew);
  float rs = 0.0f;
#pragma unroll
  for (int t = 0; t < 2; ++t)
#pragma unroll
    for (int r = 0; r < 8; ++r) {
      pv[t][r] = __expf(pv[t][r] - mnew);
      rs += pv[t][r];
    }
  rs += __shfl_xor(rs, 16, 32);
  l_run = l_run * alpha + rs;
  m_run = mnew;
#pragma unroll
  for (int dt = 0; dt < 4; ++dt)
#pragma unroll
    for (int r = 0; r < 8; ++r) oacc[dt][r] *= alpha;

  // ---- pack P^T (32 keys x 16 queries) into f16 B layout, reg-only ----
  float lowh[8], highh[8];
#pragma unroll
  for (int r = 0; r < 8; ++r) {
    const float own = hi ? pv[1][r] : pv[0][r];   // my tile-hi values
    const float snd = hi ? pv[0][r] : pv[1][r];   // partner's tile values
    const float rec = __shfl_xor(snd, 16, 32);
    lowh[r]  = hi ? rec : own;                    // keys 16*hi + 0..7
    highh[r] = hi ? own : rec;                    // keys 16*hi + 8..15
  }
  v16h bp;
#pragma unroll
  for (int j = 0; j < 4; ++j) {
    bp[2 * j]     = (_Float16)lowh[2 * j];
    bp[2 * j + 1] = (_Float16)lowh[2 * j + 1];
  }
#pragma unroll
  for (int j = 4; j < 8; ++j) {
    bp[2 * j]     = (_Float16)highh[2 * (j - 4)];
    bp[2 * j + 1] = (_Float16)highh[2 * (j - 4) + 1];
  }

  // ---- O^T += V^T x P^T: VTG A-layout, 8 contiguous keys per b128 ----
  const _Float16* vp = vt + (unsigned)nl * 1024u + (unsigned)jbase + 16u * (unsigned)hi;
#pragma unroll
  for (int dt = 0; dt < 4; ++dt) {
    U16h av;
    av.h[0] = *(const v8h*)(vp + dt * 16384);       // dim tile: +dt*16*1024
    av.h[1] = *(const v8h*)(vp + dt * 16384 + 8);   // keys +8
    oacc[dt] = __builtin_amdgcn_wmma_f32_16x16x32_f16(false, av.v, false, bp,
                                                      (short)0, oacc[dt], false, false);
  }
}

// ---------------------------------------------------------------------------
// Fast path: flash attention reading prepacked f16 KG/VTG (b128 loads only).
// First GEMM : S^T (keys x queries) = KG(A)  x Q^T(B)  -> lane = query
// Second GEMM: O^T (dims x queries) = VTG(A) x P^T(B)  -> lane = query
// ---------------------------------------------------------------------------
__global__ __launch_bounds__(128, 1) void dilated_flash_wmma_packed(
    const float* __restrict__ Q,
    const _Float16* __restrict__ ws,
    float* __restrict__ O,
    const int* __restrict__ causal_flag)
{
  const int lane = threadIdx.x & 31;
  const int wv   = threadIdx.x >> 5;
  const int wid  = blockIdx.x * 4 + wv;       // 0..3071, 1024 per group

  const int g    = wid >> 10;                 // head group 0,1,2
  const int rem  = wid & 1023;
  const int sl   = 1024 << g;
  const int lqt  = 6 + g;                     // log2(q-tiles per segment)
  const int seg  = rem >> (lqt + 2);
  const int rem2 = rem & ((4 << lqt) - 1);
  const int hg   = rem2 >> lqt;               // head in group
  const int head = (g << 2) + hg;
  const int qt   = rem2 & ((1 << lqt) - 1);

  const int nl = lane & 15;                   // query lane / A-row lane
  const int hi = lane >> 4;                   // half-wave id

  const int is_causal = causal_flag[0];
  const int seg_base  = seg * sl;
  const int hoff      = head * kD;

  const unsigned baseHalves = (g == 0) ? 0u : ((g == 1) ? kOffG1 : kOffG2);
  const unsigned tile = baseHalves + (unsigned)(seg * 4 + hg) * kTile;
  const _Float16* kg = ws + tile;
  const _Float16* vt = ws + kKGHalves + tile;

  // ---- Q^T (64 x 16 queries) -> WMMA B layout (from fp32, once) ----
  v16h bq[2];
  {
    const float* qp = Q + (seg_base + qt * 16 + nl) * kRow + hoff;
#pragma unroll
    for (int c = 0; c < 2; ++c) {
#pragma unroll
      for (int j = 0; j < 8; ++j) {
        const int dim = c * 32 + 2 * j + 8 * hi + ((j >= 4) ? 8 : 0);
        const float2 f = *(const float2*)(qp + dim);
        bq[c][2 * j]     = (_Float16)f.x;
        bq[c][2 * j + 1] = (_Float16)f.y;
      }
    }
  }

  v8f oacc[4];
#pragma unroll
  for (int dt = 0; dt < 4; ++dt)
#pragma unroll
    for (int r = 0; r < 8; ++r) oacc[dt][r] = 0.0f;
  float m_run = -1.0e30f;
  float l_run = 0.0f;

  const int qrow = qt * 16 + nl;              // this lane's query row (in-seg)

  // Fully-unmasked blocks: 32*kb+31 <= qt*16  <=>  kb < qt>>1 (causal),
  // or all blocks when non-causal. Exactly one masked diagonal block remains.
  const int kb_full = is_causal ? (qt >> 1) : (sl >> 5);
  for (int kb = 0; kb < kb_full; ++kb)
    flash_step_packed<false>(kg, vt, kb, nl, hi, qrow, bq, oacc, m_run, l_run);
  if (is_causal)                               // wave-uniform branch
    flash_step_packed<true>(kg, vt, kb_full, nl, hi, qrow, bq, oacc, m_run, l_run);

  // ---- normalize and write out: 8 consecutive dims per lane per tile ----
  const float inv = 1.0f / l_run;
  float* op = O + (seg_base + qt * 16 + nl) * kRow + hoff;
#pragma unroll
  for (int dt = 0; dt < 4; ++dt) {
    float4 w0, w1;
    w0.x = oacc[dt][0] * inv; w0.y = oacc[dt][1] * inv;
    w0.z = oacc[dt][2] * inv; w0.w = oacc[dt][3] * inv;
    w1.x = oacc[dt][4] * inv; w1.y = oacc[dt][5] * inv;
    w1.z = oacc[dt][6] * inv; w1.w = oacc[dt][7] * inv;
    const int base = dt * 16 + 8 * hi;               // dims dt*16+8*hi+0..7
    *(float4*)(op + base)     = w0;
    *(float4*)(op + base + 4) = w1;
  }
}

// ---------------------------------------------------------------------------
// Fallback (ws too small): direct-from-fp32 version.
// ---------------------------------------------------------------------------
__global__ __launch_bounds__(128, 1) void dilated_flash_wmma_direct(
    const float* __restrict__ Q,
    const float* __restrict__ K,
    const float* __restrict__ V,
    float* __restrict__ O,
    const int* __restrict__ causal_flag)
{
  const int lane = threadIdx.x & 31;
  const int wv   = threadIdx.x >> 5;
  const int wid  = blockIdx.x * 4 + wv;

  const int g    = wid >> 10;
  const int rem  = wid & 1023;
  const int sl   = 1024 << g;
  const int lqt  = 6 + g;
  const int seg  = rem >> (lqt + 2);
  const int rem2 = rem & ((4 << lqt) - 1);
  const int head = (g << 2) + (rem2 >> lqt);
  const int qt   = rem2 & ((1 << lqt) - 1);

  const int nl = lane & 15;
  const int hi = lane >> 4;

  const int   is_causal = causal_flag[0];
  const float scale     = 0.125f;
  const int   seg_base  = seg * sl;
  const int   hoff      = head * kD;

  v16h bq[2];
  {
    const float* qp = Q + (seg_base + qt * 16 + nl) * kRow + hoff;
#pragma unroll
    for (int c = 0; c < 2; ++c)
#pragma unroll
      for (int j = 0; j < 8; ++j) {
        const int dim = c * 32 + 2 * j + 8 * hi + ((j >= 4) ? 8 : 0);
        const float2 f = *(const float2*)(qp + dim);
        bq[c][2 * j]     = (_Float16)f.x;
        bq[c][2 * j + 1] = (_Float16)f.y;
      }
  }

  v8f oacc[4];
#pragma unroll
  for (int dt = 0; dt < 4; ++dt)
#pragma unroll
    for (int r = 0; r < 8; ++r) oacc[dt][r] = 0.0f;
  float m_run = -1.0e30f;
  float l_run = 0.0f;

  const int qrow = qt * 16 + nl;
  const int kb_count = is_causal ? (((qt * 16 + 15) >> 5) + 1) : (sl >> 5);

  for (int kb = 0; kb < kb_count; ++kb) {
    float pv[2][8];
#pragma unroll
    for (int t = 0; t < 2; ++t) {
      const int keyA = kb * 32 + t * 16 + nl;
      const int gpos = ((keyA & 1023) << g) + g;
      const float* kp = K + (seg_base + gpos) * kRow + hoff;
      v8f s;
#pragma unroll
      for (int r = 0; r < 8; ++r) s[r] = 0.0f;
#pragma unroll
      for (int c = 0; c < 2; ++c) {
        v16h ak;
#pragma unroll
        for (int j = 0; j < 8; ++j) {
          const int kl = 2 * j + 8 * hi + ((j >= 4) ? 8 : 0);
          const float2 f = *(const float2*)(kp + c * 32 + kl);
          ak[2 * j]     = (_Float16)f.x;
          ak[2 * j + 1] = (_Float16)f.y;
        }
        s = __builtin_amdgcn_wmma_f32_16x16x32_f16(false, ak, false, bq[c],
                                                   (short)0, s, false, false);
      }
#pragma unroll
      for (int r = 0; r < 8; ++r) {
        const int key = kb * 32 + t * 16 + r + 8 * hi;
        float val = s[r] * scale;
        if (is_causal && (key > qrow)) val = -1.0e30f;
        pv[t][r] = val;
      }
    }

    float mx = pv[0][0];
#pragma unroll
    for (int r = 1; r < 8; ++r) mx = fmaxf(mx, pv[0][r]);
#pragma unroll
    for (int r = 0; r < 8; ++r) mx = fmaxf(mx, pv[1][r]);
    mx = fmaxf(mx, __shfl_xor(mx, 16, 32));
    const float mnew  = fmaxf(m_run, mx);
    const float alpha = __expf(m_run - mnew);
    float rs = 0.0f;
#pragma unroll
    for (int t = 0; t < 2; ++t)
#pragma unroll
      for (int r = 0; r < 8; ++r) {
        pv[t][r] = __expf(pv[t][r] - mnew);
        rs += pv[t][r];
      }
    rs += __shfl_xor(rs, 16, 32);
    l_run = l_run * alpha + rs;
    m_run = mnew;
#pragma unroll
    for (int dt = 0; dt < 4; ++dt)
#pragma unroll
      for (int r = 0; r < 8; ++r) oacc[dt][r] *= alpha;

    float lowh[8], highh[8];
#pragma unroll
    for (int r = 0; r < 8; ++r) {
      const float own = hi ? pv[1][r] : pv[0][r];
      const float snd = hi ? pv[0][r] : pv[1][r];
      const float rec = __shfl_xor(snd, 16, 32);
      lowh[r]  = hi ? rec : own;
      highh[r] = hi ? own : rec;
    }
    v16h bp;
#pragma unroll
    for (int j = 0; j < 4; ++j) {
      bp[2 * j]     = (_Float16)lowh[2 * j];
      bp[2 * j + 1] = (_Float16)lowh[2 * j + 1];
    }
#pragma unroll
    for (int j = 4; j < 8; ++j) {
      bp[2 * j]     = (_Float16)highh[2 * (j - 4)];
      bp[2 * j + 1] = (_Float16)highh[2 * (j - 4) + 1];
    }

#pragma unroll
    for (int dt = 0; dt < 4; ++dt) {
      v16h av;
#pragma unroll
      for (int j = 0; j < 8; ++j) {
        const int kl   = 2 * j + 16 * hi;
        const int key0 = kb * 32 + kl;
        const int gp0  = ((key0 & 1023) << g) + g;
        const int gp1  = (((key0 + 1) & 1023) << g) + g;
        av[2 * j]     = (_Float16)V[(seg_base + gp0) * kRow + hoff + dt * 16 + nl];
        av[2 * j + 1] = (_Float16)V[(seg_base + gp1) * kRow + hoff + dt * 16 + nl];
      }
      oacc[dt] = __builtin_amdgcn_wmma_f32_16x16x32_f16(false, av, false, bp,
                                                        (short)0, oacc[dt], false, false);
    }
  }

  const float inv = 1.0f / l_run;
  float* op = O + (seg_base + qt * 16 + nl) * kRow + hoff;
#pragma unroll
  for (int dt = 0; dt < 4; ++dt) {
    float4 w0, w1;
    w0.x = oacc[dt][0] * inv; w0.y = oacc[dt][1] * inv;
    w0.z = oacc[dt][2] * inv; w0.w = oacc[dt][3] * inv;
    w1.x = oacc[dt][4] * inv; w1.y = oacc[dt][5] * inv;
    w1.z = oacc[dt][6] * inv; w1.w = oacc[dt][7] * inv;
    const int base = dt * 16 + 8 * hi;
    *(float4*)(op + base)     = w0;
    *(float4*)(op + base + 4) = w1;
  }
}

extern "C" void kernel_launch(void* const* d_in, const int* in_sizes, int n_in,
                              void* d_out, int out_size, void* d_ws, size_t ws_size,
                              hipStream_t stream) {
  (void)in_sizes; (void)n_in; (void)out_size;
  const float* q = (const float*)d_in[0];
  const float* k = (const float*)d_in[1];
  const float* v = (const float*)d_in[2];
  const int*   c = (const int*)d_in[3];
  const size_t need = (size_t)kWsHalves * sizeof(_Float16);   // 7 MB
  if (ws_size >= need) {
    // 2*917504 element-pairs, one per thread.
    prepack_kv<<<2u * kPairsPerSide / 256u, 256, 0, stream>>>(
        k, v, (_Float16*)d_ws);
    dilated_flash_wmma_packed<<<768, 128, 0, stream>>>(
        q, (const _Float16*)d_ws, (float*)d_out, c);
  } else {
    dilated_flash_wmma_direct<<<768, 128, 0, stream>>>(
        q, k, v, (float*)d_out, c);
  }
}
